// TransformerEncoder_default_19593640805002
// MI455X (gfx1250) — matrix operand order repaired and checked by reading, not compiled, not run on hardware
//
#include <hip/hip_runtime.h>
#include <hip/hip_bf16.h>
#include <math.h>

// ---------------------------------------------------------------------------
// Transformer encoder forward for MI455X (gfx1250), wave32 + WMMA f16.
// All linear layers are y = x @ W^T (+bias); attention is batched per (b,h).
// Every GEMM runs through one generic wave-tiled v_wmma_f32_16x16x32_f16
// kernel; each wave computes a 16(M) x 64(N) tile, K stepped by 32.
// ---------------------------------------------------------------------------

typedef _Float16 half8  __attribute__((ext_vector_type(8)));
typedef _Float16 half16 __attribute__((ext_vector_type(16)));
typedef float    float8 __attribute__((ext_vector_type(8)));

#define BB   16
#define SS   512
#define DD   512
#define HH   8
#define DHD  64
#define FFD  2048
#define LL   4
#define ROWS (BB*SS)   /* 8192 */
#define KPAD 96        /* input size 80 padded to multiple of 32 */

struct GemmP {
  const _Float16* A;   // [M,K] row-major, leading dim lda
  const _Float16* B;   // [N,K] row-major (weights / keys / vT), leading dim ldb
  const float*    bias;   // [N] or nullptr
  float*          Cf;     // f32 output or nullptr
  _Float16*       Ch;     // f16 output or nullptr
  int  M, N, K, lda, ldb, ldc;
  long sA, sB;            // per-batch element strides for A and B
  long cS0, cS1;          // C offset = (bz/Hdiv)*cS0 + (bz%Hdiv)*cS1
  int  Hdiv, relu, addPos;
  float alpha;
};

__global__ __launch_bounds__(256) void gemm_wmma_f16(GemmP p) {
  const int wave  = threadIdx.x >> 5;
  const int lane  = threadIdx.x & 31;
  const int mBase = (blockIdx.y * 4 + (wave >> 1)) * 16;
  const int nBase = (blockIdx.x * 2 + (wave & 1)) * 64;
  if (mBase >= p.M || nBase >= p.N) return;      // uniform per wave
  const int l16 = lane & 15;
  const int hi  = lane >> 4;
  const int bz  = blockIdx.z;

  const _Float16* Am = p.A + (size_t)bz * p.sA;
  const _Float16* Bm = p.B + (size_t)bz * p.sB;
  // A fragment: row M = mBase+l16; 16 contiguous K starting at (lane/16)*16
  const _Float16* aRow = Am + (size_t)(mBase + l16) * p.lda + (hi << 4);
  // B fragment: col N = nBase+16t+l16; 16 contiguous K starting at (lane/16)*16
  const _Float16* bRow[4];
#pragma unroll
  for (int t = 0; t < 4; ++t)
    bRow[t] = Bm + (size_t)(nBase + t * 16 + l16) * p.ldb + (hi << 4);

  float8 acc[4] = {};
  for (int k0 = 0; k0 < p.K; k0 += 32) {
    __builtin_prefetch(aRow + k0 + 64, 0, 1);        // global_prefetch_b8
    half8 alo = *(const half8*)(aRow + k0);
    half8 ahi = *(const half8*)(aRow + k0 + 8);
    half16 a  = __builtin_shufflevector(alo, ahi, 0,1,2,3,4,5,6,7,
                                        8,9,10,11,12,13,14,15);
#pragma unroll
    for (int t = 0; t < 4; ++t) {
      half16 b = *(const half16*)(bRow[t] + k0);
      acc[t] = __builtin_amdgcn_wmma_f32_16x16x32_f16(
          /*neg_a=*/false, a, /*neg_b=*/false, b,
          /*c_mod=*/(short)0, acc[t], /*reuse_a=*/false, /*reuse_b=*/false);
    }
  }

  const long cOff = (long)(bz / p.Hdiv) * p.cS0 + (long)(bz % p.Hdiv) * p.cS1;
  const int row0 = mBase + hi * 8;
#pragma unroll
  for (int t = 0; t < 4; ++t) {
    const int   col = nBase + t * 16 + l16;
    const float bv  = p.bias ? p.bias[col] : 0.0f;
    float pdiv = 0.0f;
    if (p.addPos) pdiv = __expf(-(float)(col & ~1) * 0.0179889458f); // ln(1e4)/512
#pragma unroll
    for (int r = 0; r < 8; ++r) {
      const int row = row0 + r;
      float v = acc[t][r] * p.alpha + bv;
      if (p.addPos) {
        const float ph = (float)(row & (SS - 1)) * pdiv;   // s = row % S
        v += (col & 1) ? __cosf(ph) : __sinf(ph);
      }
      if (p.relu) v = fmaxf(v, 0.0f);
      const long idx = cOff + (long)row * p.ldc + col;
      if (p.Cf) p.Cf[idx] = v;
      if (p.Ch) p.Ch[idx] = (_Float16)v;
    }
  }
}

// f32 [N,K] -> f16 [N,Kp] with zero padding for k >= K
__global__ void cvt_pad_f16(const float* __restrict__ src,
                            _Float16* __restrict__ dst, int N, int K, int Kp) {
  size_t i = (size_t)blockIdx.x * blockDim.x + threadIdx.x;
  if (i >= (size_t)N * Kp) return;
  int kp = (int)(i % Kp);
  size_t n = i / Kp;
  dst[i] = (_Float16)((kp < K) ? src[n * K + kp] : 0.0f);
}

// qkv16 [B,S,3D] -> q16/k16 [B,H,S,64] and vT16 [B,H,64,S]
__global__ void split_qkv(const _Float16* __restrict__ qkv,
                          _Float16* __restrict__ q, _Float16* __restrict__ k,
                          _Float16* __restrict__ vT) {
  size_t i = (size_t)blockIdx.x * blockDim.x + threadIdx.x;
  if (i >= (size_t)ROWS * 3 * DD) return;
  int    c  = (int)(i % (3 * DD));
  size_t bs = i / (3 * DD);
  int    s  = (int)(bs % SS);
  int    b  = (int)(bs / SS);
  int which = c / DD, d = c % DD, h = d / DHD, dd = d % DHD;
  _Float16 v = qkv[i];
  size_t bh = (size_t)b * HH + h;
  if      (which == 0) q [(bh * SS + s) * DHD + dd] = v;
  else if (which == 1) k [(bh * SS + s) * DHD + dd] = v;
  else                 vT[(bh * DHD + dd) * SS + s] = v;
}

// One block (128 threads) per attention row; masked softmax -> f16 probs
__global__ __launch_bounds__(128) void softmax_mask(
    const float* __restrict__ scores, const int* __restrict__ lens,
    _Float16* __restrict__ attn) {
  __shared__ float red[128];
  const int rowId = blockIdx.x;                 // bh*S + q
  const int tid   = threadIdx.x;
  const int b     = rowId / (HH * SS);
  const int len   = lens[b];
  const float* src = scores + (size_t)rowId * SS;
  _Float16*    dst = attn   + (size_t)rowId * SS;

  float x[4];
  float m = -1e30f;
#pragma unroll
  for (int j = 0; j < 4; ++j) {
    int c = tid + j * 128;
    x[j] = (c < len) ? src[c] : -1e30f;
    m = fmaxf(m, x[j]);
  }
  red[tid] = m; __syncthreads();
  for (int s2 = 64; s2 > 0; s2 >>= 1) {
    if (tid < s2) red[tid] = fmaxf(red[tid], red[tid + s2]);
    __syncthreads();
  }
  m = red[0]; __syncthreads();

  float sum = 0.0f;
#pragma unroll
  for (int j = 0; j < 4; ++j) { x[j] = __expf(x[j] - m); sum += x[j]; }
  red[tid] = sum; __syncthreads();
  for (int s2 = 64; s2 > 0; s2 >>= 1) {
    if (tid < s2) red[tid] += red[tid + s2];
    __syncthreads();
  }
  const float inv = 1.0f / red[0];
#pragma unroll
  for (int j = 0; j < 4; ++j) {
    int c = tid + j * 128;
    dst[c] = (_Float16)((c < len) ? x[j] * inv : 0.0f);
  }
}

// h = LayerNorm(res + x) ; also emit f16 copy for the next GEMM
__global__ __launch_bounds__(128) void add_layernorm(
    const float* __restrict__ res, const float* __restrict__ x,
    const float* __restrict__ g, const float* __restrict__ bb,
    float* __restrict__ out, _Float16* __restrict__ out16) {
  __shared__ float r1[128], r2[128];
  const size_t base = (size_t)blockIdx.x * DD;
  const int tid = threadIdx.x;
  float v[4], s = 0.0f, sq = 0.0f;
#pragma unroll
  for (int j = 0; j < 4; ++j) {
    int c = tid + j * 128;
    v[j] = res[base + c] + x[base + c];
    s += v[j]; sq += v[j] * v[j];
  }
  r1[tid] = s; r2[tid] = sq; __syncthreads();
  for (int s2 = 64; s2 > 0; s2 >>= 1) {
    if (tid < s2) { r1[tid] += r1[tid + s2]; r2[tid] += r2[tid + s2]; }
    __syncthreads();
  }
  const float mean = r1[0] * (1.0f / DD);
  const float var  = r2[0] * (1.0f / DD) - mean * mean;
  const float rstd = rsqrtf(var + 1e-5f);
#pragma unroll
  for (int j = 0; j < 4; ++j) {
    int c = tid + j * 128;
    float y = (v[j] - mean) * rstd * g[c] + bb[c];
    out[base + c] = y;
    out16[base + c] = (_Float16)y;
  }
}

// pooled[b,d] = mean over valid s of h[b,s,d]; lens appended bit-exact
__global__ void pool_mean(const float* __restrict__ h,
                          const int* __restrict__ lens, float* __restrict__ out) {
  int i = blockIdx.x * blockDim.x + threadIdx.x;   // over B*D
  if (i >= BB * DD) return;
  int b = i / DD, d = i % DD;
  int len = lens[b];
  float s = 0.0f;
  for (int t = 0; t < len; ++t) s += h[((size_t)b * SS + t) * DD + d];
  out[i] = s / (float)len;
  if (d == 0) ((int*)out)[BB * DD + b] = len;
}

// ---------------------------------------------------------------------------
static void launch_gemm(hipStream_t st,
                        const _Float16* A, int lda, long sA,
                        const _Float16* B, int ldb, long sB,
                        const float* bias, float* Cf, _Float16* Ch, int ldc,
                        long cS0, long cS1, int Hdiv, int batch,
                        int M, int N, int K, float alpha, int relu, int addPos) {
  GemmP p;
  p.A = A; p.B = B; p.bias = bias; p.Cf = Cf; p.Ch = Ch;
  p.M = M; p.N = N; p.K = K; p.lda = lda; p.ldb = ldb; p.ldc = ldc;
  p.sA = sA; p.sB = sB; p.cS0 = cS0; p.cS1 = cS1;
  p.Hdiv = Hdiv; p.relu = relu; p.addPos = addPos; p.alpha = alpha;
  dim3 grid((N + 127) / 128, (M + 63) / 64, batch);
  gemm_wmma_f16<<<grid, dim3(256), 0, st>>>(p);
}

extern "C" void kernel_launch(void* const* d_in, const int* in_sizes, int n_in,
                              void* d_out, int out_size, void* d_ws, size_t ws_size,
                              hipStream_t stream) {
  const float* x     = (const float*)d_in[0];
  const int*   lens  = (const int*)  d_in[1];
  const float* We    = (const float*)d_in[2];
  const float* be    = (const float*)d_in[3];
  const float* Wqkv  = (const float*)d_in[4];
  const float* bqkv  = (const float*)d_in[5];
  const float* Wo    = (const float*)d_in[6];
  const float* bo    = (const float*)d_in[7];
  const float* ln1g  = (const float*)d_in[8];
  const float* ln1b  = (const float*)d_in[9];
  const float* W1    = (const float*)d_in[10];
  const float* b1    = (const float*)d_in[11];
  const float* W2    = (const float*)d_in[12];
  const float* b2    = (const float*)d_in[13];
  const float* ln2g  = (const float*)d_in[14];
  const float* ln2b  = (const float*)d_in[15];
  float* out = (float*)d_out;

  char* ws = (char*)d_ws;
  size_t off = 0;
  auto alloc = [&](size_t bytes) -> void* {
    off = (off + 255) & ~(size_t)255;
    void* p = ws + off; off += bytes; return p;
  };

  _Float16* We16   = (_Float16*)alloc((size_t)DD * KPAD * 2);
  _Float16* Wqkv16 = (_Float16*)alloc((size_t)LL * 3 * DD * DD * 2);
  _Float16* Wo16   = (_Float16*)alloc((size_t)LL * DD * DD * 2);
  _Float16* W116   = (_Float16*)alloc((size_t)LL * FFD * DD * 2);
  _Float16* W216   = (_Float16*)alloc((size_t)LL * DD * FFD * 2);
  _Float16* x16    = (_Float16*)alloc((size_t)ROWS * KPAD * 2);
  float*    h      = (float*)   alloc((size_t)ROWS * DD * 4);
  _Float16* h16    = (_Float16*)alloc((size_t)ROWS * DD * 2);
  _Float16* qkv16  = (_Float16*)alloc((size_t)ROWS * 3 * DD * 2);
  _Float16* q16    = (_Float16*)alloc((size_t)ROWS * DD * 2);
  _Float16* k16    = (_Float16*)alloc((size_t)ROWS * DD * 2);
  _Float16* vT16   = (_Float16*)alloc((size_t)ROWS * DD * 2);
  float*    scores = (float*)   alloc((size_t)BB * HH * SS * SS * 4);
  _Float16* attn16 = (_Float16*)alloc((size_t)BB * HH * SS * SS * 2);
  _Float16* ctx16  = (_Float16*)alloc((size_t)ROWS * DD * 2);
  float*    tmp    = (float*)   alloc((size_t)ROWS * DD * 4);
  _Float16* ff116  = (_Float16*)alloc((size_t)ROWS * FFD * 2);
  (void)ws_size; (void)in_sizes; (void)n_in; (void)out_size;

  auto cvt = [&](const float* s, _Float16* d, int N, int K, int Kp) {
    size_t tot = (size_t)N * Kp;
    cvt_pad_f16<<<dim3((unsigned)((tot + 255) / 256)), dim3(256), 0, stream>>>(s, d, N, K, Kp);
  };
  // weight / input conversion (f32 -> f16, K padded where needed)
  cvt(We,   We16,   DD,          80,  KPAD);
  cvt(Wqkv, Wqkv16, LL * 3 * DD, DD,  DD);
  cvt(Wo,   Wo16,   LL * DD,     DD,  DD);
  cvt(W1,   W116,   LL * FFD,    DD,  DD);
  cvt(W2,   W216,   LL * DD,     FFD, FFD);
  cvt(x,    x16,    ROWS,        80,  KPAD);

  // Embedding: h = x @ We^T + be + posenc   (writes f32 + f16)
  launch_gemm(stream, x16, KPAD, 0, We16, KPAD, 0, be, h, h16, DD,
              0, 0, 1, 1, ROWS, DD, KPAD, 1.0f, 0, /*addPos=*/1);

  const int BH = BB * HH;
  for (int l = 0; l < LL; ++l) {
    // QKV projection -> f16 [B*S, 3D]
    launch_gemm(stream, h16, DD, 0, Wqkv16 + (size_t)l * 3 * DD * DD, DD, 0,
                bqkv + (size_t)l * 3 * DD, nullptr, qkv16, 3 * DD,
                0, 0, 1, 1, ROWS, 3 * DD, DD, 1.0f, 0, 0);
    {
      size_t tot = (size_t)ROWS * 3 * DD;
      split_qkv<<<dim3((unsigned)((tot + 255) / 256)), dim3(256), 0, stream>>>(
          qkv16, q16, k16, vT16);
    }
    // scores = (q @ k^T) / sqrt(dh), batched over 128 (b,h)
    launch_gemm(stream, q16, DHD, (long)SS * DHD, k16, DHD, (long)SS * DHD,
                nullptr, scores, nullptr, SS,
                (long)SS * SS, 0, 1, BH, SS, SS, DHD, 0.125f, 0, 0);
    softmax_mask<<<dim3(BH * SS), dim3(128), 0, stream>>>(scores, lens, attn16);
    // ctx = attn @ v  -> scattered into [B,S,D] layout (col offset h*64)
    launch_gemm(stream, attn16, SS, (long)SS * SS, vT16, SS, (long)DHD * SS,
                nullptr, nullptr, ctx16, DD,
                (long)SS * DD, (long)DHD, HH, BH, SS, DHD, SS, 1.0f, 0, 0);
    // output projection
    launch_gemm(stream, ctx16, DD, 0, Wo16 + (size_t)l * DD * DD, DD, 0,
                bo + (size_t)l * DD, tmp, nullptr, DD,
                0, 0, 1, 1, ROWS, DD, DD, 1.0f, 0, 0);
    add_layernorm<<<dim3(ROWS), dim3(128), 0, stream>>>(
        h, tmp, ln1g + (size_t)l * DD, ln1b + (size_t)l * DD, h, h16);
    // FFN1 (+ReLU) -> f16 [B*S, F]
    launch_gemm(stream, h16, DD, 0, W116 + (size_t)l * FFD * DD, DD, 0,
                b1 + (size_t)l * FFD, nullptr, ff116, FFD,
                0, 0, 1, 1, ROWS, FFD, DD, 1.0f, /*relu=*/1, 0);
    // FFN2
    launch_gemm(stream, ff116, FFD, 0, W216 + (size_t)l * DD * FFD, FFD, 0,
                b2 + (size_t)l * DD, tmp, nullptr, DD,
                0, 0, 1, 1, ROWS, DD, FFD, 1.0f, 0, 0);
    add_layernorm<<<dim3(ROWS), dim3(128), 0, stream>>>(
        h, tmp, ln2g + (size_t)l * DD, ln2b + (size_t)l * DD, h, h16);
  }

  pool_mean<<<dim3((BB * DD + 255) / 256), dim3(256), 0, stream>>>(h, lens, out);
}
